// Matchmaker_context_v1_42674795053889
// MI455X (gfx1250) — compile-verified, matrix operand order checked by reading, not run on hardware
//
#include <hip/hip_runtime.h>
#include <stdint.h>

// ---------------- constants ----------------
#define BB   128
#define QQ   30
#define DD   200
#define QP   32
#define DP   208
#define EE   300
#define E2   600
#define KC0  608        // padded K for level-0 cosine (600 -> 608)
#define KNG  320        // padded K for ngram cosine (300 -> 320)
#define NOUT 300        // conv out channels
#define NPAD 304        // padded conv out channels (19 n-tiles)
#define NK   11         // number of RBF kernels

typedef __attribute__((ext_vector_type(16))) _Float16 v16h;
typedef __attribute__((ext_vector_type(8)))  _Float16 v8h;
typedef __attribute__((ext_vector_type(8)))  float    v8f;

__constant__ float c_mu[NK]   = {1.0f, 0.9f, 0.7f, 0.5f, 0.3f, 0.1f,
                                 -0.1f, -0.3f, -0.5f, -0.7f, -0.9f};
// 1/(2*sigma^2): sigma0=1e-4 -> 5e7 ; others sigma=0.1 -> 50
__constant__ float c_i2s2[NK] = {5.0e7f, 50.f, 50.f, 50.f, 50.f, 50.f,
                                 50.f, 50.f, 50.f, 50.f, 50.f};

// ---------------- WMMA helpers (CDNA5 16x16x32 f16 -> f32) ----------------
// A/B fragment layout (ISA 7.12.2, 16-bit 16x32): lane m=lane&15, half=lane>>4,
// element e -> k = (e>>3)*16 + half*8 + (e&7)  => two contiguous 8-half loads.
__device__ __forceinline__ v16h load_frag_rm(const _Float16* p, int ld) {
  int lane = threadIdx.x & 31;
  int r = lane & 15;
  int h = lane >> 4;
  const _Float16* base = p + (size_t)r * ld + h * 8;
  v8h lo = *(const v8h*)(base);
  v8h hi = *(const v8h*)(base + 16);
  return __builtin_shufflevector(lo, hi, 0, 1, 2, 3, 4, 5, 6, 7,
                                 8, 9, 10, 11, 12, 13, 14, 15);
}

__device__ __forceinline__ v8f wmma_f16(v16h a, v16h b, v8f c) {
  return __builtin_amdgcn_wmma_f32_16x16x32_f16(false, a, false, b,
                                                (short)0, c, false, false);
}

// ---------------- misc ----------------
__global__ void zero32_kernel(uint32_t* p, size_t n) {
  for (size_t i = (size_t)blockIdx.x * blockDim.x + threadIdx.x; i < n;
       i += (size_t)gridDim.x * blockDim.x)
    p[i] = 0u;
}

// ---------------- encoder stage 1: posenc + FF + LN + QKV proj ----------------
__global__ __launch_bounds__(128)
void enc_stage1_kernel(const float* __restrict__ emb,
                       const float* __restrict__ W1, const float* __restrict__ b1,
                       const float* __restrict__ W2, const float* __restrict__ b2,
                       const float* __restrict__ lg, const float* __restrict__ lb,
                       const float* __restrict__ Wc, const float* __restrict__ bc,
                       float* __restrict__ ff_out, float* __restrict__ comb_out,
                       int T) {
  int t = blockIdx.x, b = blockIdx.y, tid = threadIdx.x;
  __shared__ float xs[EE], hs[100], ffs[EE], red[128];
  const float log_inc = 0.06181436f;  // log(10000)/149
  const float* xrow = emb + ((size_t)b * T + t) * EE;
  for (int e = tid; e < EE; e += 128) {
    float pe;
    if (e < 150) pe = sinf((float)t * expf(-(float)e * log_inc));
    else         pe = cosf((float)t * expf(-(float)(e - 150) * log_inc));
    xs[e] = xrow[e] + pe;
  }
  __syncthreads();
  for (int j = tid; j < 100; j += 128) {
    float acc = b1[j];
    for (int e = 0; e < EE; ++e) acc += xs[e] * W1[e * 100 + j];
    hs[j] = fmaxf(acc, 0.f);
  }
  __syncthreads();
  for (int e = tid; e < EE; e += 128) {
    float acc = b2[e];
    for (int j = 0; j < 100; ++j) acc += hs[j] * W2[j * EE + e];
    ffs[e] = acc + xs[e];
  }
  __syncthreads();
  float p = 0.f;
  for (int e = tid; e < EE; e += 128) p += ffs[e];
  red[tid] = p; __syncthreads();
  for (int s = 64; s > 0; s >>= 1) { if (tid < s) red[tid] += red[tid + s]; __syncthreads(); }
  float mean = red[0] * (1.f / 300.f);
  __syncthreads();
  p = 0.f;
  for (int e = tid; e < EE; e += 128) { float d = ffs[e] - mean; p += d * d; }
  red[tid] = p; __syncthreads();
  for (int s = 64; s > 0; s >>= 1) { if (tid < s) red[tid] += red[tid + s]; __syncthreads(); }
  float stdv = sqrtf(red[0] * (1.f / 299.f));  // ddof=1
  __syncthreads();
  float* fo = ff_out + ((size_t)b * T + t) * EE;
  for (int e = tid; e < EE; e += 128) {
    float y = lg[e] * (ffs[e] - mean) / (stdv + 1e-6f) + lb[e];
    fo[e] = y;
    ffs[e] = y;
  }
  __syncthreads();
  float* co = comb_out + ((size_t)b * T + t) * 96;
  for (int c = tid; c < 96; c += 128) {
    float acc = bc[c];
    for (int e = 0; e < EE; ++e) acc += ffs[e] * Wc[e * 96 + c];
    co[c] = acc;
  }
}

// ---------------- encoder stage 2: per-head attention (dh = 1) ----------------
__global__ __launch_bounds__(256)
void enc_attn_kernel(const float* __restrict__ comb, const float* __restrict__ mask,
                     float* __restrict__ attout, int T) {
  int h = blockIdx.x, b = blockIdx.y, tid = threadIdx.x;
  __shared__ float kk[DP], vv[DP], mk[DP];
  const float* cb = comb + (size_t)b * T * 96;
  for (int t = tid; t < T; t += 256) {
    kk[t] = cb[t * 96 + 32 + h];
    vv[t] = cb[t * 96 + 64 + h];
    mk[t] = mask[b * T + t];
  }
  __syncthreads();
  for (int t = tid; t < T; t += 256) {
    float qv = cb[t * 96 + h] * (1.f / 3.f);  // scale = sqrt(300//32) = 3
    float mx = -1e30f;
    for (int k2 = 0; k2 < T; ++k2) { float z = qv * kk[k2] * mk[k2]; mx = fmaxf(mx, z); }
    float se = 0.f, sem = 0.f, sev = 0.f;
    for (int k2 = 0; k2 < T; ++k2) {
      float z = qv * kk[k2] * mk[k2];
      float e = expf(z - mx);
      se += e;
      float em = e * mk[k2];
      sem += em;
      sev += em * vv[k2];
    }
    // softmax*m renormalized: out = sev / (sem + 1e-13*se)
    attout[((size_t)b * T + t) * 32 + h] = sev / (sem + 1e-13f * se);
  }
}

// ---------------- encoder stage 3: out proj + LN + f16 concat (+normalized) ----
__global__ __launch_bounds__(128)
void enc_stage3_kernel(const float* __restrict__ emb, const float* __restrict__ attout,
                       const float* __restrict__ Wo, const float* __restrict__ bo,
                       const float* __restrict__ lg, const float* __restrict__ lb,
                       const float* __restrict__ ff,
                       _Float16* __restrict__ ccat, _Float16* __restrict__ ccatn,
                       int T, int Mp) {
  int t = blockIdx.x, b = blockIdx.y, tid = threadIdx.x;
  __shared__ float as[32], cc[E2], red[128];
  if (tid < 32) as[tid] = attout[((size_t)b * T + t) * 32 + tid];
  __syncthreads();
  const float* fr = ff + ((size_t)b * T + t) * EE;
  for (int e = tid; e < EE; e += 128) {
    float acc = bo[e];
    for (int a = 0; a < 32; ++a) acc += as[a] * Wo[a * EE + e];
    cc[EE + e] = acc + fr[e];
  }
  __syncthreads();
  float p = 0.f;
  for (int e = tid; e < EE; e += 128) p += cc[EE + e];
  red[tid] = p; __syncthreads();
  for (int s = 64; s > 0; s >>= 1) { if (tid < s) red[tid] += red[tid + s]; __syncthreads(); }
  float mean = red[0] * (1.f / 300.f);
  __syncthreads();
  p = 0.f;
  for (int e = tid; e < EE; e += 128) { float d = cc[EE + e] - mean; p += d * d; }
  red[tid] = p; __syncthreads();
  for (int s = 64; s > 0; s >>= 1) { if (tid < s) red[tid] += red[tid + s]; __syncthreads(); }
  float stdv = sqrtf(red[0] * (1.f / 299.f));
  __syncthreads();
  const float* er = emb + ((size_t)b * T + t) * EE;
  for (int e = tid; e < EE; e += 128) {
    float y = lg[e] * (cc[EE + e] - mean) / (stdv + 1e-6f) + lb[e];
    cc[EE + e] = y;
    cc[e] = er[e];
  }
  __syncthreads();
  p = 0.f;
  for (int e = tid; e < E2; e += 128) { float v = cc[e]; p += v * v; }
  red[tid] = p; __syncthreads();
  for (int s = 64; s > 0; s >>= 1) { if (tid < s) red[tid] += red[tid + s]; __syncthreads(); }
  float inv = 1.f / (sqrtf(red[0]) + 1e-13f);
  _Float16* o1 = ccat + ((size_t)b * Mp + t) * KC0;
  _Float16* o2 = ccatn + ((size_t)b * Mp + t) * KC0;
  for (int e = tid; e < E2; e += 128) {
    o1[e] = (_Float16)cc[e];
    o2[e] = (_Float16)(cc[e] * inv);
  }
}

// ---------------- doc lengths ----------------
__global__ __launch_bounds__(256)
void doclen_kernel(const float* __restrict__ dm, float* __restrict__ dlen) {
  int b = blockIdx.x, tid = threadIdx.x;
  __shared__ float red[256];
  float p = 0.f;
  for (int d = tid; d < DD; d += 256) p += dm[b * DD + d];
  red[tid] = p; __syncthreads();
  for (int s = 128; s > 0; s >>= 1) { if (tid < s) red[tid] += red[tid + s]; __syncthreads(); }
  if (tid == 0) dlen[b] = red[0];
}

// ---------------- conv weight convert: (300,600,i) f32 -> (304,Kp) f16 N-major -
__global__ __launch_bounds__(256)
void wconv_kernel(const float* __restrict__ W, _Float16* __restrict__ Wt,
                  int win, int Kp) {
  size_t n = (size_t)NPAD * Kp;
  int Kreal = E2 * win;
  for (size_t idx = (size_t)blockIdx.x * 256 + threadIdx.x; idx < n;
       idx += (size_t)gridDim.x * 256) {
    int o = (int)(idx / Kp);
    int kap = (int)(idx - (size_t)o * Kp);
    float v = 0.f;
    if (o < NOUT && kap < Kreal) {
      int j = kap / E2;
      int c = kap - j * E2;
      v = W[((size_t)o * E2 + c) * win + j];  // W[o][c][j]
    }
    Wt[idx] = (_Float16)v;
  }
}

// ---------------- cosine GEMM: C[b] = A(Mp,Kp) * B(Np,Kp)^T via WMMA ----------
__global__ __launch_bounds__(256)
void cosine_wmma_kernel(const _Float16* __restrict__ A, const _Float16* __restrict__ Bm,
                        float* __restrict__ C, int Mp, int Np, int Kp) {
  int b = blockIdx.z;
  int wave = threadIdx.x >> 5;
  int mt = blockIdx.x;
  int nt = blockIdx.y * 8 + wave;
  if (nt * 16 >= Np) return;  // wave-uniform
  const _Float16* Ab = A + (size_t)b * Mp * Kp + (size_t)mt * 16 * Kp;
  const _Float16* Bb = Bm + (size_t)b * Np * Kp + (size_t)nt * 16 * Kp;
  v8f acc = {};
  for (int k = 0; k < Kp; k += 32) {
    v16h af = load_frag_rm(Ab + k, Kp);
    v16h bf = load_frag_rm(Bb + k, Kp);
    acc = wmma_f16(af, bf, acc);
  }
  int lane = threadIdx.x & 31;
  int n = lane & 15, hm = lane >> 4;
  float* Cb = C + (size_t)b * Mp * Np + (size_t)mt * 16 * Np + nt * 16 + n;
  #pragma unroll
  for (int v = 0; v < 8; ++v) {
    int m = hm * 8 + v;
    Cb[(size_t)m * Np] = acc[v];
  }
}

// ---------------- conv GEMM (im2col in LDS): Y = relu(X_win * Wt^T + bias) -----
__global__ __launch_bounds__(256)
void conv_wmma_kernel(const _Float16* __restrict__ X,   // (B, Mp, 608) raw concat
                      const _Float16* __restrict__ Wt,  // (304, Kp) f16
                      const float* __restrict__ bias,   // (300)
                      float* __restrict__ Y,            // (B, Mp, 300)
                      int T, int Mp, int win, int Kp) {
  __shared__ _Float16 Alds[16 * 64];
  int b = blockIdx.z;
  int mt = blockIdx.x;
  int wave = threadIdx.x >> 5;
  int lane = threadIdx.x & 31;
  int nt = blockIdx.y * 8 + wave;
  bool valid = (nt * 16) < NOUT;  // wave-uniform
  int Kreal = E2 * win;
  const _Float16* Xb = X + (size_t)b * Mp * KC0;
  v8f acc = {};
  int base = threadIdx.x * 4;
  int r = base >> 6;
  int kk = base & 63;
  int t0 = mt * 16 + r;
  for (int kc = 0; kc < Kp; kc += 64) {
    __syncthreads();
    #pragma unroll
    for (int e = 0; e < 4; ++e) {
      int kap = kc + kk + e;
      _Float16 v = (_Float16)0.f;
      if (kap < Kreal) {
        int j = kap / E2;
        int c = kap - j * E2;
        int tt = t0 + j;
        if (tt < T) v = Xb[(size_t)tt * KC0 + c];
      }
      Alds[r * 64 + kk + e] = v;
    }
    __syncthreads();
    if (valid) {
      const _Float16* Wb = Wt + (size_t)nt * 16 * Kp + kc;
      v16h a0 = load_frag_rm((const _Float16*)Alds, 64);
      v16h b0 = load_frag_rm(Wb, Kp);
      acc = wmma_f16(a0, b0, acc);
      v16h a1 = load_frag_rm((const _Float16*)Alds + 32, 64);
      v16h b1 = load_frag_rm(Wb + 32, Kp);
      acc = wmma_f16(a1, b1, acc);
    }
  }
  if (valid) {
    int n = lane & 15, hm = lane >> 4;
    int o = nt * 16 + n;
    if (o < NOUT) {
      float bv = bias[o];
      #pragma unroll
      for (int v = 0; v < 8; ++v) {
        int m = hm * 8 + v;
        int t = mt * 16 + m;
        if (t < T) Y[((size_t)b * Mp + t) * NOUT + o] = fmaxf(acc[v] + bv, 0.f);
      }
    }
  }
}

// ---------------- row L2-normalize conv output to padded f16 (Mp,320) ----------
__global__ __launch_bounds__(128)
void rownorm_kernel(const float* __restrict__ Y, _Float16* __restrict__ N,
                    int T, int Mp) {
  int t = blockIdx.x, b = blockIdx.y, tid = threadIdx.x;
  __shared__ float red[128];
  _Float16* out = N + ((size_t)b * Mp + t) * KNG;
  if (t >= T) {
    for (int o = tid; o < KNG; o += 128) out[o] = (_Float16)0.f;
    return;
  }
  const float* row = Y + ((size_t)b * Mp + t) * NOUT;
  float ss = 0.f;
  for (int o = tid; o < NOUT; o += 128) { float v = row[o]; ss += v * v; }
  red[tid] = ss; __syncthreads();
  for (int s = 64; s > 0; s >>= 1) { if (tid < s) red[tid] += red[tid + s]; __syncthreads(); }
  float inv = 1.f / (sqrtf(red[0]) + 1e-13f);
  for (int o = tid; o < KNG; o += 128)
    out[o] = (o < NOUT) ? (_Float16)(row[o] * inv) : (_Float16)0.f;
}

// ---------------- kernel pooling: tanh + RBF + masked log-sums -----------------
__global__ __launch_bounds__(256)
void pool_kernel(const float* __restrict__ mm, const float* __restrict__ qm,
                 const float* __restrict__ dm, const float* __restrict__ dlen,
                 float* __restrict__ sums, float* __restrict__ means, int lvl) {
  int q = blockIdx.x, b = blockIdx.y, tid = threadIdx.x;
  __shared__ float rbuf[256];
  __shared__ float tot[NK];
  float qmv = qm[b * QQ + q];
  const float* row = mm + ((size_t)b * QP + q) * DP;
  const float* dmb = dm + b * DD;
  float kq[NK];
  #pragma unroll
  for (int k = 0; k < NK; ++k) kq[k] = 0.f;
  for (int d = tid; d < DD; d += 256) {
    float tv = tanhf(row[d]);
    float msk = qmv * dmb[d];
    #pragma unroll
    for (int k = 0; k < NK; ++k) {
      float df = tv - c_mu[k];
      kq[k] += expf(-df * df * c_i2s2[k]) * msk;
    }
  }
  for (int k = 0; k < NK; ++k) {
    rbuf[tid] = kq[k]; __syncthreads();
    for (int s = 128; s > 0; s >>= 1) { if (tid < s) rbuf[tid] += rbuf[tid + s]; __syncthreads(); }
    if (tid == 0) tot[k] = rbuf[0];
    __syncthreads();
  }
  if (tid == 0) {
    float dl = dlen[b];
    for (int k = 0; k < NK; ++k) {
      float kqv = tot[k];
      atomicAdd(&sums[b * (4 * NK) + lvl * NK + k], logf(fmaxf(kqv, 1e-10f)) * qmv);
      atomicAdd(&means[b * (4 * NK) + lvl * NK + k], logf(fmaxf(kqv / dl, 1e-10f)) * qmv);
    }
  }
}

// ---------------- final scoring ----------------
__global__ __launch_bounds__(128)
void score_kernel(const float* __restrict__ sums, const float* __restrict__ means,
                  const float* __restrict__ lsW, const float* __restrict__ lsb,
                  const float* __restrict__ lmW, const float* __restrict__ lmb,
                  const float* __restrict__ lcW, float* __restrict__ out) {
  int b = threadIdx.x;
  if (b >= BB) return;
  float ls = lsb[0], lm = lmb[0];
  for (int j = 0; j < 4 * NK; ++j) {
    ls += sums[b * (4 * NK) + j] * lsW[j];
    lm += means[b * (4 * NK) + j] * lmW[j];
  }
  out[b] = ls * lcW[0] + lm * lcW[1];
}

// ---------------- host launcher ----------------
extern "C" void kernel_launch(void* const* d_in, const int* in_sizes, int n_in,
                              void* d_out, int out_size, void* d_ws, size_t ws_size,
                              hipStream_t stream) {
  (void)in_sizes; (void)n_in; (void)out_size; (void)ws_size;
  const float* qe   = (const float*)d_in[0];
  const float* de   = (const float*)d_in[1];
  const float* qm   = (const float*)d_in[2];
  const float* dm   = (const float*)d_in[3];
  const float* ffW1 = (const float*)d_in[4];
  const float* ffb1 = (const float*)d_in[5];
  const float* ffW2 = (const float*)d_in[6];
  const float* ffb2 = (const float*)d_in[7];
  const float* flg  = (const float*)d_in[8];
  const float* flb  = (const float*)d_in[9];
  const float* aWc  = (const float*)d_in[10];
  const float* abc  = (const float*)d_in[11];
  const float* aWo  = (const float*)d_in[12];
  const float* abo  = (const float*)d_in[13];
  const float* lng  = (const float*)d_in[14];
  const float* lnb  = (const float*)d_in[15];
  const float* convW[3] = {(const float*)d_in[16], (const float*)d_in[18], (const float*)d_in[20]};
  const float* convB[3] = {(const float*)d_in[17], (const float*)d_in[19], (const float*)d_in[21]};
  const float* lsW = (const float*)d_in[22];
  const float* lsb = (const float*)d_in[23];
  const float* lmW = (const float*)d_in[24];
  const float* lmb = (const float*)d_in[25];
  const float* lcW = (const float*)d_in[26];
  float* out = (float*)d_out;

  const int wins[3] = {2, 3, 4};
  const int kps[3]  = {1216, 1856, 2432};  // 600*i rounded up to 64

  // workspace layout (deterministic bump allocator, 256B aligned)
  size_t off = 0;
  char* wsb = (char*)d_ws;
  auto alloc = [&](size_t bytes) -> char* {
    char* p = wsb + off;
    off += (bytes + 255) & ~(size_t)255;
    return p;
  };
  _Float16* ccat_q  = (_Float16*)alloc((size_t)BB * QP * KC0 * 2);
  _Float16* ccat_d  = (_Float16*)alloc((size_t)BB * DP * KC0 * 2);
  _Float16* ccatn_q = (_Float16*)alloc((size_t)BB * QP * KC0 * 2);
  _Float16* ccatn_d = (_Float16*)alloc((size_t)BB * DP * KC0 * 2);
  _Float16* Wt[3];
  for (int l = 0; l < 3; ++l) Wt[l] = (_Float16*)alloc((size_t)NPAD * kps[l] * 2);
  _Float16* ngq = (_Float16*)alloc((size_t)BB * QP * KNG * 2);
  _Float16* ngd = (_Float16*)alloc((size_t)BB * DP * KNG * 2);
  float* mm    = (float*)alloc((size_t)BB * QP * DP * 4);
  float* sums  = (float*)alloc((size_t)BB * 4 * NK * 4);
  float* means = (float*)alloc((size_t)BB * 4 * NK * 4);
  float* dlen  = (float*)alloc((size_t)BB * 4);
  // scratch (encoder first, then overlaid by conv outputs)
  char* scratch = alloc((size_t)BB * DD * EE * 4 + (size_t)BB * DD * 96 * 4 +
                        (size_t)BB * DD * 32 * 4);
  float* ffS   = (float*)scratch;
  float* combS = (float*)(scratch + (size_t)BB * DD * EE * 4);
  float* attS  = (float*)(scratch + (size_t)BB * DD * EE * 4 + (size_t)BB * DD * 96 * 4);
  float* convQ = (float*)scratch;                                    // (B,32,300)
  float* convD = (float*)(scratch + (size_t)BB * QP * NOUT * 4);     // (B,208,300)

  // ---- zero init (pads of normalized concat + pooled accumulators) ----
  zero32_kernel<<<2048, 256, 0, stream>>>((uint32_t*)ccatn_q, (size_t)BB * QP * KC0 / 2);
  zero32_kernel<<<2048, 256, 0, stream>>>((uint32_t*)ccatn_d, (size_t)BB * DP * KC0 / 2);
  zero32_kernel<<<64, 256, 0, stream>>>((uint32_t*)sums, (size_t)BB * 4 * NK);
  zero32_kernel<<<64, 256, 0, stream>>>((uint32_t*)means, (size_t)BB * 4 * NK);

  // ---- encoder: queries, then docs (scratch reused) ----
  enc_stage1_kernel<<<dim3(QQ, BB), 128, 0, stream>>>(qe, ffW1, ffb1, ffW2, ffb2,
                                                      flg, flb, aWc, abc, ffS, combS, QQ);
  enc_attn_kernel<<<dim3(32, BB), 256, 0, stream>>>(combS, qm, attS, QQ);
  enc_stage3_kernel<<<dim3(QQ, BB), 128, 0, stream>>>(qe, attS, aWo, abo, lng, lnb,
                                                      ffS, ccat_q, ccatn_q, QQ, QP);
  enc_stage1_kernel<<<dim3(DD, BB), 128, 0, stream>>>(de, ffW1, ffb1, ffW2, ffb2,
                                                      flg, flb, aWc, abc, ffS, combS, DD);
  enc_attn_kernel<<<dim3(32, BB), 256, 0, stream>>>(combS, dm, attS, DD);
  enc_stage3_kernel<<<dim3(DD, BB), 128, 0, stream>>>(de, attS, aWo, abo, lng, lnb,
                                                      ffS, ccat_d, ccatn_d, DD, DP);
  doclen_kernel<<<BB, 256, 0, stream>>>(dm, dlen);

  // ---- conv weights -> f16 (N x K row-major, padded) ----
  for (int l = 0; l < 3; ++l)
    wconv_kernel<<<1024, 256, 0, stream>>>(convW[l], Wt[l], wins[l], kps[l]);

  // ---- level 0: cosine on concat (K=608) + pooling ----
  cosine_wmma_kernel<<<dim3(QP / 16, 2, BB), 256, 0, stream>>>(ccatn_q, ccatn_d, mm,
                                                               QP, DP, KC0);
  pool_kernel<<<dim3(QQ, BB), 256, 0, stream>>>(mm, qm, dm, dlen, sums, means, 0);

  // ---- levels 1..3: conv (WMMA GEMM) + normalize + cosine + pooling ----
  for (int l = 0; l < 3; ++l) {
    conv_wmma_kernel<<<dim3(QP / 16, 3, BB), 256, 0, stream>>>(ccat_q, Wt[l], convB[l],
                                                               convQ, QQ, QP, wins[l], kps[l]);
    conv_wmma_kernel<<<dim3(DP / 16, 3, BB), 256, 0, stream>>>(ccat_d, Wt[l], convB[l],
                                                               convD, DD, DP, wins[l], kps[l]);
    rownorm_kernel<<<dim3(QP, BB), 128, 0, stream>>>(convQ, ngq, QQ, QP);
    rownorm_kernel<<<dim3(DP, BB), 128, 0, stream>>>(convD, ngd, DD, DP);
    cosine_wmma_kernel<<<dim3(QP / 16, 2, BB), 256, 0, stream>>>(ngq, ngd, mm, QP, DP, KNG);
    pool_kernel<<<dim3(QQ, BB), 256, 0, stream>>>(mm, qm, dm, dlen, sums, means, l + 1);
  }

  // ---- final linears -> score ----
  score_kernel<<<1, 128, 0, stream>>>(sums, means, lsW, lsb, lmW, lmb, lcW, out);
}
